// HeteGraphRecNodeAggregator_67430986547810
// MI455X (gfx1250) — matrix operand order repaired and checked by reading, not compiled, not run on hardware
//
#include <hip/hip_runtime.h>

typedef __attribute__((ext_vector_type(16))) __bf16 v16bf;
typedef __attribute__((ext_vector_type(8)))  float  v8f;

#define D_DIM   256
#define H_DIM   256
#define K_NB    32
#define TILE_M  16
#define THREADS 256
#define NWAVES  8

__global__ __launch_bounds__(THREADS, 1)
void hgr_fused_kernel(const float* __restrict__ embed,
                      const int*   __restrict__ node_ids,
                      const int*   __restrict__ nb_ids,
                      const float* __restrict__ wSelf,
                      const float* __restrict__ wNb,
                      const float* __restrict__ bias,
                      float*       __restrict__ out,
                      int Ntotal, int ntiles)
{
    // 2 x 128KB transposed bf16 weights + ping-pong 2 x (2 x 8KB) pooled activations = 288KB LDS
    __shared__ __bf16 sWt[2][H_DIM * D_DIM];
    __shared__ __bf16 sA [2][2][TILE_M * D_DIM];   // [buf][gemm][m*d]

    const int tid  = threadIdx.x;
    const int wave = tid >> 5;
    const int lane = tid & 31;

    // ---- one-time: load both weights to LDS, transposed (Wt[h][d] = W[d][h]), bf16 ----
    for (int idx = tid; idx < D_DIM * H_DIM; idx += THREADS) {
        int d = idx >> 8;       // W row
        int h = idx & 255;      // W col (consecutive tid -> consecutive h: coalesced reads)
        sWt[0][h * D_DIM + d] = (__bf16)wSelf[idx];
        sWt[1][h * D_DIM + d] = (__bf16)wNb[idx];
    }

    const int g      = wave >> 2;         // 0 = self GEMM, 1 = neighbor GEMM
    const int hbase  = (wave & 3) * 4;    // 4 h-tiles (of 16 cols) per wave
    const int mrow   = lane & 15;
    const int khalfA = (lane >> 4) * 16;  // A-frag: upper lane half holds K+16
    const int khalfB = (lane >> 4) * 16;  // B-frag: upper lane half holds K+16
    const int rowoff = (lane >> 4) * 8;   // C-frag: upper lane half holds M+8

    // ---- gather + mean-pool one tile into a ping-pong buffer ----
    auto pool = [&](int tile, int buf) {
        const int n0 = tile * TILE_M;
        for (int nn = wave; nn < TILE_M; nn += NWAVES) {   // wave handles nodes nn, nn+8
            const int node = n0 + nn;
            __bf16* aSelf = &sA[buf][0][nn * D_DIM];
            __bf16* aNb   = &sA[buf][1][nn * D_DIM];
            if (node < Ntotal) {
                const int self = node_ids[node];
                const float* srow = embed + (size_t)self * D_DIM;
                #pragma unroll
                for (int j = 0; j < D_DIM / 32; ++j)
                    aSelf[lane + 32 * j] = (__bf16)srow[lane + 32 * j];

                float acc[D_DIM / 32];
                #pragma unroll
                for (int j = 0; j < D_DIM / 32; ++j) acc[j] = 0.f;
                const int* nbrow = nb_ids + (size_t)node * K_NB;  // wave-uniform -> scalar loads
                for (int k = 0; k < K_NB; ++k) {
                    const float* erow = embed + (size_t)nbrow[k] * D_DIM;
                    #pragma unroll
                    for (int j = 0; j < D_DIM / 32; ++j)
                        acc[j] += erow[lane + 32 * j];
                }
                #pragma unroll
                for (int j = 0; j < D_DIM / 32; ++j)
                    aNb[lane + 32 * j] = (__bf16)(acc[j] * (1.f / K_NB));
            } else {
                #pragma unroll
                for (int j = 0; j < D_DIM / 32; ++j) {
                    aSelf[lane + 32 * j] = (__bf16)0.f;
                    aNb  [lane + 32 * j] = (__bf16)0.f;
                }
            }
        }
    };

    // ---- prologue: pool first tile, cover weight staging with the same barrier ----
    pool(blockIdx.x, 0);
    __syncthreads();

    for (int i = 0; ; ++i) {
        const int tile = blockIdx.x + i * (int)gridDim.x;
        const int next = tile + (int)gridDim.x;
        const int n0   = tile * TILE_M;

        // 1) kick off pooling of the NEXT tile (global loads overlap this tile's WMMAs)
        if (next < ntiles) pool(next, (i + 1) & 1);

        // 2) GEMM this tile: 4 output 16x16 tiles per wave, 8 x wmma 16x16x32 bf16
        v8f c0 = {}, c1 = {}, c2 = {}, c3 = {};
        const __bf16* Ag = &sA[i & 1][g][0];
        const __bf16* Wg = &sWt[g][0];
        #pragma unroll
        for (int ks = 0; ks < D_DIM / 32; ++ks) {
            // A fragment: 16 contiguous K per lane (k = 16*half + s), same order as B
            union { v16bf v; unsigned u[8]; } a;
            const int abase = mrow * D_DIM + ks * 32;
            #pragma unroll
            for (int j = 0; j < 8; ++j) {
                const int klo = khalfA + 2 * j;
                a.u[j] = *reinterpret_cast<const unsigned*>(Ag + abase + klo);
            }
            // B fragments: lane = column, 16 contiguous K from transposed weight
            const int bbase = ks * 32 + khalfB;
            v16bf b0 = *reinterpret_cast<const v16bf*>(Wg + ((hbase + 0) * 16 + mrow) * D_DIM + bbase);
            v16bf b1 = *reinterpret_cast<const v16bf*>(Wg + ((hbase + 1) * 16 + mrow) * D_DIM + bbase);
            v16bf b2 = *reinterpret_cast<const v16bf*>(Wg + ((hbase + 2) * 16 + mrow) * D_DIM + bbase);
            v16bf b3 = *reinterpret_cast<const v16bf*>(Wg + ((hbase + 3) * 16 + mrow) * D_DIM + bbase);

            c0 = __builtin_amdgcn_wmma_f32_16x16x32_bf16(false, a.v, false, b0, (short)0, c0, false, false);
            c1 = __builtin_amdgcn_wmma_f32_16x16x32_bf16(false, a.v, false, b1, (short)0, c1, false, false);
            c2 = __builtin_amdgcn_wmma_f32_16x16x32_bf16(false, a.v, false, b2, (short)0, c2, false, false);
            c3 = __builtin_amdgcn_wmma_f32_16x16x32_bf16(false, a.v, false, b3, (short)0, c3, false, false);
        }

        // 3) epilogue: bias + relu + store (C layout: M = r + 8*lanehi, N = lane&15)
        v8f cc[4] = {c0, c1, c2, c3};
        #pragma unroll
        for (int t = 0; t < 4; ++t) {
            const int col = g * H_DIM + (hbase + t) * 16 + mrow;
            const float bv = bias[col];
            #pragma unroll
            for (int r = 0; r < 8; ++r) {
                const int node = n0 + rowoff + r;
                if (node < Ntotal) {
                    float v = cc[t][r] + bv;
                    out[(size_t)node * (2 * H_DIM) + col] = v > 0.f ? v : 0.f;
                }
            }
        }

        if (next >= ntiles) break;
        // single barrier/tile: publishes pool(next) to all, and fences buf reuse
        __syncthreads();
    }
}

extern "C" void kernel_launch(void* const* d_in, const int* in_sizes, int n_in,
                              void* d_out, int out_size, void* d_ws, size_t ws_size,
                              hipStream_t stream) {
    const float* embed    = (const float*)d_in[0];
    const int*   node_ids = (const int*)  d_in[1];
    const int*   nb_ids   = (const int*)  d_in[2];
    const float* wSelf    = (const float*)d_in[3];
    const float* wNb      = (const float*)d_in[4];
    const float* bias     = (const float*)d_in[5];
    float*       out      = (float*)d_out;

    const int N      = in_sizes[1];
    const int ntiles = (N + TILE_M - 1) / TILE_M;
    const int grid   = ntiles < 256 ? ntiles : 256;   // persistent WGs: weights loaded once each

    hipLaunchKernelGGL(hgr_fused_kernel, dim3(grid), dim3(THREADS), 0, stream,
                       embed, node_ids, nb_ids, wSelf, wNb, bias, out, N, ntiles);
}
